// Block_81046032876263
// MI455X (gfx1250) — matrix operand order repaired and checked by reading, not compiled, not run on hardware
//
#include <hip/hip_runtime.h>

typedef __attribute__((ext_vector_type(16))) _Float16 v16h;
typedef __attribute__((ext_vector_type(8)))  _Float16 v8h;
typedef __attribute__((ext_vector_type(4)))  _Float16 v4h;
typedef __attribute__((ext_vector_type(8)))  float    v8f;
typedef __attribute__((ext_vector_type(4)))  float    vf4;

#define BDIM  4
#define TDIM  512
#define EDIM  1024
#define HDIM  16
#define DDIM  64
#define MROWS (BDIM*TDIM)   // 2048

__device__ __forceinline__ v8f wmma16(v16h a, v16h b, v8f c) {
    // (neg_a, A, neg_b, B, c_mod, C, reuse_a, reuse_b)
    return __builtin_amdgcn_wmma_f32_16x16x32_f16(false, a, false, b, (short)0, c, false, false);
}

__device__ __forceinline__ unsigned lds_off(const void* p) {
    // generic (flat) pointer to LDS: low 32 bits are the LDS byte offset
    return (unsigned)(size_t)p;
}

// ---------------------------------------------------------------------------
// Weight pre-pack: f32 (wblk-addressed) -> f16 tile-transposed layout.
//   source element (k,n) at (n/wblk)*K*wblk + k*wblk + (n%wblk)
//   dest: tile t=(n/64)*(K/32)+(k/32), within tile [n%64][k%32] contiguous.
// Each 64x32 tile is a contiguous 4 KB block = exactly one GEMM B stage.
// ---------------------------------------------------------------------------
__global__ void __launch_bounds__(256)
pack_w(const float* __restrict__ W, int wblk, int K, int N, _Float16* __restrict__ out)
{
    int e = blockIdx.x * 256 + threadIdx.x;   // e < K*N
    int k = e / N, n = e % N;
    size_t wi = (size_t)(n / wblk) * (size_t)K * (size_t)wblk
              + (size_t)k * (size_t)wblk + (size_t)(n % wblk);
    size_t oi = ((size_t)(n >> 6) * (size_t)(K >> 5) + (size_t)(k >> 5)) * 2048
              + (size_t)(n & 63) * 32 + (size_t)(k & 31);
    out[oi] = (_Float16)W[wi];
}

// ---------------------------------------------------------------------------
// GEMM: C[M,N] = act( concatK(A0,A1)[M,K] * Wp + bias )
//   A: f32, register-pipelined (b128 loads of tile i+1 overlap compute of i),
//      converted to f16 and staged to LDS as b64 stores.
//   Wp: pre-packed f16 tiles, double-buffered global_load_async_to_lds_b128
//      DMA (ASYNCcnt) overlapping the WMMA burst.
// Block: 128x64 C tile, 128 threads (4 waves), each wave 32x64 (8 WMMAs/step).
// ---------------------------------------------------------------------------
__global__ void __launch_bounds__(128)
gemm_wmma(const float* __restrict__ A0, const float* __restrict__ A1,
          int ksplit, int lda0, int lda1,
          const _Float16* __restrict__ Wp,
          const float* __restrict__ bias, int relu,
          float* __restrict__ Cf, _Float16* __restrict__ Ch,
          int N, int K)
{
    __shared__ __attribute__((aligned(16))) _Float16 As[128][40];   // [row][k]
    __shared__ __attribute__((aligned(16))) _Float16 Bp[2][2048];   // packed tiles

    const int tid  = threadIdx.x;
    const int wv   = tid >> 5;
    const int lane = tid & 31;
    const int l16  = lane & 15;
    const int hi   = lane >> 4;
    const int n0   = blockIdx.x * 64;
    const int m0   = blockIdx.y * 128;
    const int iters = K >> 5;

    v8f acc[2][4] = {};
    vf4 areg[8];

    auto issueB = [&](int it) {
        const _Float16* gsrc = Wp + ((size_t)(n0 >> 6) * (size_t)iters + (size_t)it) * 2048;
        unsigned lbase = lds_off(&Bp[it & 1][0]);
        #pragma unroll
        for (int t = 0; t < 2; ++t) {
            int e = tid + t * 128;                 // 16B chunk id (0..255)
            const _Float16* src = gsrc + e * 8;
            unsigned dst = lbase + e * 16;
            asm volatile("global_load_async_to_lds_b128 %0, %1, off"
                         :: "v"(dst), "v"(src) : "memory");
        }
    };
    auto loadA = [&](int it) {                     // 8 x b128 global loads
        const int k0 = it << 5;
        #pragma unroll
        for (int t = 0; t < 8; ++t) {
            int chunk = tid + t * 128;             // 0..1023 (16B chunks)
            int r = chunk >> 3, c4 = chunk & 7;
            int gk = k0 + c4 * 4;
            areg[t] = (gk < ksplit)
                        ? *(const vf4*)(A0 + (size_t)(m0 + r) * lda0 + gk)
                        : *(const vf4*)(A1 + (size_t)(m0 + r) * lda1 + (gk - ksplit));
        }
    };

    loadA(0);
    issueB(0);

    for (int it = 0; it < iters; ++it) {
        // --- store staged A regs (tile it) as f16 b64 writes ---
        #pragma unroll
        for (int t = 0; t < 8; ++t) {
            int chunk = tid + t * 128;
            int r = chunk >> 3, c4 = chunk & 7;
            v4h hh;
            #pragma unroll
            for (int j = 0; j < 4; ++j) hh[j] = (_Float16)areg[t][j];
            *(v4h*)&As[r][c4 * 4] = hh;
        }
        // --- prefetch tile it+1 (A regs + B DMA) overlapping compute ---
        if (it + 1 < iters) {
            loadA(it + 1);
            issueB(it + 1);
            asm volatile("s_wait_asynccnt 2" ::: "memory");
        } else {
            asm volatile("s_wait_asynccnt 0" ::: "memory");
        }
        __syncthreads();

        const _Float16* bt = &Bp[it & 1][0];

        // --- 2 A fragments (rows 32*wv+{0..15,16..31}) ---
        v16h a0, a1, b[4];
        {
            const int kbase = hi ? 8 : 0;
            const int r0 = (wv << 5) + l16;
            union { vf4 f; _Float16 h[8]; } u0, u1, u2, u3;
            u0.f = *(const vf4*)&As[r0][kbase];
            u1.f = *(const vf4*)&As[r0][kbase + 16];
            u2.f = *(const vf4*)&As[r0 + 16][kbase];
            u3.f = *(const vf4*)&As[r0 + 16][kbase + 16];
            #pragma unroll
            for (int i = 0; i < 8; ++i) {
                a0[i] = u0.h[i]; a0[i + 8] = u1.h[i];
                a1[i] = u2.h[i]; a1[i + 8] = u3.h[i];
            }
        }
        // --- preload all 4 B fragments, then 8-WMMA burst ---
        #pragma unroll
        for (int nt = 0; nt < 4; ++nt) {
            const int col = (nt << 4) + l16;
            const int kb  = hi ? 8 : 0;
            union { vf4 f; _Float16 h[8]; } u0, u1;
            u0.f = *(const vf4*)&bt[col * 32 + kb];
            u1.f = *(const vf4*)&bt[col * 32 + kb + 16];
            #pragma unroll
            for (int i = 0; i < 8; ++i) { b[nt][i] = u0.h[i]; b[nt][i + 8] = u1.h[i]; }
        }
        #pragma unroll
        for (int nt = 0; nt < 4; ++nt) {
            acc[0][nt] = wmma16(a0, b[nt], acc[0][nt]);
            acc[1][nt] = wmma16(a1, b[nt], acc[1][nt]);
        }
        __syncthreads();
    }

    // --- epilogue ---
    #pragma unroll
    for (int r = 0; r < 2; ++r) {
        #pragma unroll
        for (int nt = 0; nt < 4; ++nt) {
            const int col = n0 + (nt << 4) + l16;
            const float bv = bias ? bias[col] : 0.f;
            #pragma unroll
            for (int j = 0; j < 8; ++j) {
                const int row = m0 + (wv << 5) + (r << 4) + j + (hi << 3);
                float v = acc[r][nt][j] + bv;
                if (relu) v = fmaxf(v, 0.f);
                if (Ch) Ch[(size_t)row * N + col] = (_Float16)v;
                else    Cf[(size_t)row * N + col] = v;
            }
        }
    }
}

// ---------------------------------------------------------------------------
// Causal flash attention. q/k/v: f16 [B*T, H*D]; o: f32 [B*T, H*D].
// One block per (b,h,64-query tile); 4 waves x 16 queries.
// K chunks DMA'd via global_load_async_to_lds_b128 (padded per-lane dests).
// scores = q k^T (NO 1/sqrt(d) scaling, matching reference).
// ---------------------------------------------------------------------------
__global__ void __launch_bounds__(128)
attn_flash(const _Float16* __restrict__ qg, const _Float16* __restrict__ kg,
           const _Float16* __restrict__ vg, float* __restrict__ og)
{
    __shared__ __attribute__((aligned(16))) _Float16 Ks[32][72];     // [key][d]
    __shared__ __attribute__((aligned(16))) _Float16 Vt[64][40];     // [d][key]
    __shared__ __attribute__((aligned(16))) _Float16 Ps[4][16][32];  // per-wave P

    const int tid  = threadIdx.x;
    const int wv   = tid >> 5;
    const int lane = tid & 31;
    const int l16  = lane & 15;
    const int hi   = lane >> 4;

    const int qb = blockIdx.x & 7;          // T/64 = 8 query tiles
    const int bh = blockIdx.x >> 3;
    const int b  = bh >> 4;
    const int h  = bh & 15;
    const int q0  = qb * 64;
    const int qt0 = q0 + (wv << 4);

    const size_t base = (size_t)b * TDIM * EDIM + (size_t)h * DDIM;

    // Q fragments for d-chunks [0,32) and [32,64): 4 contiguous 16B loads
    const int qrow = qt0 + l16;
    v16h alo, ahi;
    {
        const _Float16* qp = qg + base + (size_t)qrow * EDIM + (hi ? 8 : 0);
        v8h q0v = *(const v8h*)(qp);
        v8h q1v = *(const v8h*)(qp + 16);
        v8h q2v = *(const v8h*)(qp + 32);
        v8h q3v = *(const v8h*)(qp + 48);
        #pragma unroll
        for (int i = 0; i < 8; ++i) {
            alo[i] = q0v[i]; alo[i + 8] = q1v[i];
            ahi[i] = q2v[i]; ahi[i + 8] = q3v[i];
        }
    }

    v8f  oacc[4] = {};
    float mrow[8], lrow[8];
    #pragma unroll
    for (int j = 0; j < 8; ++j) { mrow[j] = -3.0e38f; lrow[j] = 0.f; }

    const unsigned ksb = lds_off(&Ks[0][0]);
    const int send = q0 + 64;               // exclusive causal bound for block

    for (int s0 = 0; s0 < send; s0 += 32) {
        // --- K chunk: async DMA, 32 keys x 64 halves, padded rows (72) ---
        #pragma unroll
        for (int t = 0; t < 2; ++t) {
            int e = tid + t * 128;          // 16B chunk: key=e/8, part=e%8
            int key = e >> 3, part = e & 7;
            const _Float16* src = kg + base + (size_t)(s0 + key) * EDIM + part * 8;
            unsigned dst = ksb + key * 144 + part * 16;
            asm volatile("global_load_async_to_lds_b128 %0, %1, off"
                         :: "v"(dst), "v"(src) : "memory");
        }
        // --- V chunk: batched b128 loads, then transpose-scatter to Vt ---
        {
            v8h vreg[2];
            #pragma unroll
            for (int t = 0; t < 2; ++t) {
                int e = tid + t * 128;
                int key = e >> 3, part = e & 7;
                vreg[t] = *(const v8h*)(vg + base + (size_t)(s0 + key) * EDIM + part * 8);
            }
            #pragma unroll
            for (int t = 0; t < 2; ++t) {
                int e = tid + t * 128;
                int key = e >> 3, part = e & 7;
                #pragma unroll
                for (int i = 0; i < 8; ++i) Vt[part * 8 + i][key] = vreg[t][i];
            }
        }
        asm volatile("s_wait_asynccnt 0" ::: "memory");
        __syncthreads();

        // --- scores: two 16-key tiles, K-dim = d (2 WMMAs each) ---
        v8f sc[2];
        #pragma unroll
        for (int t = 0; t < 2; ++t) {
            const int krow = (t << 4) + l16;
            const int db   = hi ? 8 : 0;
            v16h bk0, bk1;
            union { vf4 f; _Float16 h[8]; } u0, u1, u2, u3;
            u0.f = *(const vf4*)&Ks[krow][db];
            u1.f = *(const vf4*)&Ks[krow][db + 16];
            u2.f = *(const vf4*)&Ks[krow][32 + db];
            u3.f = *(const vf4*)&Ks[krow][32 + db + 16];
            #pragma unroll
            for (int i = 0; i < 8; ++i) {
                bk0[i] = u0.h[i]; bk0[i + 8] = u1.h[i];
                bk1[i] = u2.h[i]; bk1[i + 8] = u3.h[i];
            }
            v8f z = {};
            z = wmma16(alo, bk0, z);
            sc[t] = wmma16(ahi, bk1, z);
        }

        // --- causal mask + online softmax (rows split across lane halves) ---
        #pragma unroll
        for (int j = 0; j < 8; ++j) {
            const int qrj = qt0 + j + (hi << 3);
            float s0v = sc[0][j], s1v = sc[1][j];
            if (s0 + l16 > qrj)      s0v = -3.0e38f;
            if (s0 + 16 + l16 > qrj) s1v = -3.0e38f;

            float mx = fmaxf(s0v, s1v);
            #pragma unroll
            for (int d = 1; d < 16; d <<= 1) mx = fmaxf(mx, __shfl_xor(mx, d, 32));
            const float mnew  = fmaxf(mrow[j], mx);
            const float scale = __expf(mrow[j] - mnew);
            const float p0 = __expf(s0v - mnew);
            const float p1 = __expf(s1v - mnew);
            float rs = p0 + p1;
            #pragma unroll
            for (int d = 1; d < 16; d <<= 1) rs += __shfl_xor(rs, d, 32);
            lrow[j] = lrow[j] * scale + rs;
            mrow[j] = mnew;
            #pragma unroll
            for (int nt = 0; nt < 4; ++nt) oacc[nt][j] *= scale;

            const int pr = j + (hi << 3);
            Ps[wv][pr][l16]      = (_Float16)p0;
            Ps[wv][pr][16 + l16] = (_Float16)p1;
        }
        __syncthreads();

        // --- P (16x32) x V (32x64): 4 WMMAs into o accumulators ---
        v16h ap;
        {
            const int kb = hi ? 8 : 0;
            union { vf4 f; _Float16 h[8]; } u0, u1;
            u0.f = *(const vf4*)&Ps[wv][l16][kb];
            u1.f = *(const vf4*)&Ps[wv][l16][kb + 16];
            #pragma unroll
            for (int i = 0; i < 8; ++i) { ap[i] = u0.h[i]; ap[i + 8] = u1.h[i]; }
        }
        #pragma unroll
        for (int nt = 0; nt < 4; ++nt) {
            const int col = (nt << 4) + l16;
            const int kb  = hi ? 8 : 0;
            v16h bv;
            union { vf4 f; _Float16 h[8]; } u0, u1;
            u0.f = *(const vf4*)&Vt[col][kb];
            u1.f = *(const vf4*)&Vt[col][kb + 16];
            #pragma unroll
            for (int i = 0; i < 8; ++i) { bv[i] = u0.h[i]; bv[i + 8] = u1.h[i]; }
            oacc[nt] = wmma16(ap, bv, oacc[nt]);
        }
        __syncthreads();
    }

    // --- normalize and store o (f32) ---
    #pragma unroll
    for (int nt = 0; nt < 4; ++nt) {
        #pragma unroll
        for (int j = 0; j < 8; ++j) {
            const int row  = qt0 + j + (hi << 3);
            const int dcol = (nt << 4) + l16;
            og[base + (size_t)row * EDIM + dcol] = oacc[nt][j] / lrow[j];
        }
    }
}

// ---------------------------------------------------------------------------
// Per-head LayerNorm over D=64 + residual:  xout = xin + LN(o)*g + b
// ---------------------------------------------------------------------------
__global__ void __launch_bounds__(256)
headln_res(const float* __restrict__ xin, const float* __restrict__ o,
           const float* __restrict__ g, const float* __restrict__ bb,
           float* __restrict__ xout)
{
    const int row = blockIdx.x;
    const int tid = threadIdx.x;
    const int h = tid >> 4, il = tid & 15;
    const size_t p = (size_t)row * EDIM + h * 64 + il * 4;

    vf4 v0 = *(const vf4*)(o + p);
    float s = v0[0] + v0[1] + v0[2] + v0[3];
    float s2 = v0[0]*v0[0] + v0[1]*v0[1] + v0[2]*v0[2] + v0[3]*v0[3];
    #pragma unroll
    for (int d = 1; d < 16; d <<= 1) { s += __shfl_xor(s, d, 32); s2 += __shfl_xor(s2, d, 32); }
    const float mean = s * (1.f / 64.f);
    const float var  = s2 * (1.f / 64.f) - mean * mean;
    const float inv  = rsqrtf(var + 1e-5f);

    vf4 xi = *(const vf4*)(xin + p);
    vf4 gv = *(const vf4*)(g + h * 64 + il * 4);
    vf4 bv = *(const vf4*)(bb + h * 64 + il * 4);
    vf4 out;
    #pragma unroll
    for (int j = 0; j < 4; ++j)
        out[j] = xi[j] + (v0[j] - mean) * inv * gv[j] + bv[j];
    *(vf4*)(xout + p) = out;
}

// ---------------------------------------------------------------------------
// Full-row LayerNorm over E=1024 + residual:  xout = xin + LN(hb)*g + b
// ---------------------------------------------------------------------------
__global__ void __launch_bounds__(256)
ln_res(const float* __restrict__ xin, const float* __restrict__ hb,
       const float* __restrict__ g, const float* __restrict__ bb,
       float* __restrict__ xout)
{
    __shared__ float sm[8], sv[8];
    const int row = blockIdx.x, tid = threadIdx.x;
    const size_t p = (size_t)row * EDIM;

    vf4 v0 = *(const vf4*)(hb + p + tid * 4);
    float s = v0[0] + v0[1] + v0[2] + v0[3];
    float s2 = v0[0]*v0[0] + v0[1]*v0[1] + v0[2]*v0[2] + v0[3]*v0[3];
    #pragma unroll
    for (int d = 1; d < 32; d <<= 1) { s += __shfl_xor(s, d, 32); s2 += __shfl_xor(s2, d, 32); }
    if ((tid & 31) == 0) { sm[tid >> 5] = s; sv[tid >> 5] = s2; }
    __syncthreads();
    float ts = 0.f, ts2 = 0.f;
    #pragma unroll
    for (int w = 0; w < 8; ++w) { ts += sm[w]; ts2 += sv[w]; }
    const float mean = ts * (1.f / 1024.f);
    const float var  = ts2 * (1.f / 1024.f) - mean * mean;
    const float inv  = rsqrtf(var + 1e-5f);

    vf4 xi = *(const vf4*)(xin + p + tid * 4);
    vf4 gv = *(const vf4*)(g + tid * 4);
    vf4 bv = *(const vf4*)(bb + tid * 4);
    vf4 out;
    #pragma unroll
    for (int j = 0; j < 4; ++j)
        out[j] = xi[j] + (v0[j] - mean) * inv * gv[j] + bv[j];
    *(vf4*)(xout + p + tid * 4) = out;
}

// ---------------------------------------------------------------------------
extern "C" void kernel_launch(void* const* d_in, const int* in_sizes, int n_in,
                              void* d_out, int out_size, void* d_ws, size_t ws_size,
                              hipStream_t stream)
{
    (void)in_sizes; (void)n_in; (void)out_size; (void)ws_size;
    const float* x   = (const float*)d_in[0];
    const float* wq  = (const float*)d_in[1];   // [5,H,E,D]
    const float* wk  = (const float*)d_in[2];
    const float* wvp = (const float*)d_in[3];
    const float* mg  = (const float*)d_in[4];   // [5,H,D]
    const float* mb  = (const float*)d_in[5];
    const float* w1  = (const float*)d_in[6];   // [4,E,4E]
    const float* b1  = (const float*)d_in[7];
    const float* w2  = (const float*)d_in[8];   // [4,4E,E]
    const float* b2  = (const float*)d_in[9];
    const float* fg  = (const float*)d_in[10];
    const float* fbb = (const float*)d_in[11];
    const float* ltw = (const float*)d_in[12];  // [2E,E]
    const float* ltb = (const float*)d_in[13];

    float* ws = (float*)d_ws;
    const size_t S = (size_t)MROWS * EDIM;      // one activation tensor
    // f32 region: 8*S floats
    float* s0  = ws;                            // ping
    float* s1  = ws + S;                        // pong
    float* one = ws + 2 * S;                    // out_one
    float* ob  = ws + 3 * S;                    // attn out / ff h2
    float* h1  = ws + 4 * S;                    // [M,4E] = 4S
    // f16 region
    _Float16* hbse = (_Float16*)(ws + 8 * S);
    _Float16* qh = hbse;
    _Float16* kh = hbse + S;
    _Float16* vh = hbse + 2 * S;
    const size_t W1M = (size_t)1024 * 1024;
    _Float16* pqkv = hbse + 3 * S;              // 15 x 1M halves
    _Float16* pw1  = pqkv + 15 * W1M;           // 4 x 4M
    _Float16* pw2  = pw1 + 16 * W1M;            // 4 x 4M
    _Float16* plt  = pw2 + 16 * W1M;            // 2M

    auto pack = [&](const float* W, int wblk, int K, int N, _Float16* out) {
        pack_w<<<(K * N) / 256, 256, 0, stream>>>(W, wblk, K, N, out);
    };
    const size_t WS = (size_t)HDIM * EDIM * DDIM;       // per-layer qkv stride
    for (int l = 0; l < 5; ++l) {
        pack(wq  + l * WS, 64, EDIM, EDIM, pqkv + (size_t)(l * 3 + 0) * W1M);
        pack(wk  + l * WS, 64, EDIM, EDIM, pqkv + (size_t)(l * 3 + 1) * W1M);
        pack(wvp + l * WS, 64, EDIM, EDIM, pqkv + (size_t)(l * 3 + 2) * W1M);
    }
    for (int j = 0; j < 4; ++j) {
        pack(w1 + (size_t)j * 4 * W1M, 4 * EDIM, EDIM, 4 * EDIM, pw1 + (size_t)j * 4 * W1M);
        pack(w2 + (size_t)j * 4 * W1M, EDIM, 4 * EDIM, EDIM,     pw2 + (size_t)j * 4 * W1M);
    }
    pack(ltw, EDIM, 2 * EDIM, EDIM, plt);

    auto gemm = [&](const float* A, const _Float16* Wp, const float* bias, int relu,
                    float* Cf, _Float16* Ch, int N, int K) {
        dim3 grid(N / 64, MROWS / 128);
        gemm_wmma<<<grid, 128, 0, stream>>>(A, A, K, K, K, Wp, bias, relu, Cf, Ch, N, K);
    };
    auto mh = [&](int l, const float* xin, float* xout) {
        gemm(xin, pqkv + (size_t)(l * 3 + 0) * W1M, nullptr, 0, nullptr, qh, EDIM, EDIM);
        gemm(xin, pqkv + (size_t)(l * 3 + 1) * W1M, nullptr, 0, nullptr, kh, EDIM, EDIM);
        gemm(xin, pqkv + (size_t)(l * 3 + 2) * W1M, nullptr, 0, nullptr, vh, EDIM, EDIM);
        attn_flash<<<BDIM * HDIM * (TDIM / 64), 128, 0, stream>>>(qh, kh, vh, ob);
        headln_res<<<MROWS, 256, 0, stream>>>(xin, ob, mg + l * HDIM * DDIM,
                                              mb + l * HDIM * DDIM, xout);
    };
    auto ff = [&](int j, const float* xin, float* xout) {
        gemm(xin, pw1 + (size_t)j * 4 * W1M, b1 + j * 4 * EDIM, 1, h1, nullptr, 4 * EDIM, EDIM);
        gemm(h1,  pw2 + (size_t)j * 4 * W1M, b2 + j * EDIM, 0, ob, nullptr, EDIM, 4 * EDIM);
        ln_res<<<MROWS, 256, 0, stream>>>(xin, ob, fg + j * EDIM, fbb + j * EDIM, xout);
    };

    // head_one branch: ff3(mh4(mh0(x)))
    mh(0, x, s1);
    mh(4, s1, s0);
    ff(3, s0, one);

    // encoder branch: 3x (mh, ff) then decoder (mh4, ff3)
    mh(1, x, s1);  ff(0, s1, s0);
    mh(2, s0, s1); ff(1, s1, s0);
    mh(3, s0, s1); ff(2, s1, s0);
    mh(4, s0, s1); ff(3, s1, s0);   // out_two in s0

    // final: out = concat(out_one, out_two) @ lt_w + lt_b  (K = 2E concat GEMM)
    dim3 grid(EDIM / 64, MROWS / 128);
    gemm_wmma<<<grid, 128, 0, stream>>>(one, s0, EDIM, EDIM, EDIM,
                                        plt, ltb, 0, (float*)d_out, nullptr,
                                        EDIM, 2 * EDIM);
}
